// PIKR_51213190037873
// MI455X (gfx1250) — matrix operand order repaired and checked by reading, not compiled, not run on hardware
//
#include <hip/hip_runtime.h>

#define M_DIM 8192
#define N_DIM 8192
#define D_DIM 256
#define GAMMA_F (1.0f / 256.0f)
#define BM 256            // M rows per block (8 waves x 32 rows)
#define NSPLIT 16         // N split factor for partials
#define NCHUNK (N_DIM / NSPLIT)

typedef __attribute__((ext_vector_type(16))) _Float16 v16h;
typedef __attribute__((ext_vector_type(8)))  _Float16 v8h;
typedef __attribute__((ext_vector_type(8)))  float    v8f;

// ---------------------------------------------------------------------------
// Kernel 1: fp32 -> f16 conversion + fp32 row squared-norms.
// One block per row (M+N rows), 256 threads = D elements.
// ---------------------------------------------------------------------------
__global__ __launch_bounds__(256) void cvt_norm_kernel(
    const float* __restrict__ Xq, const float* __restrict__ Xt,
    _Float16* __restrict__ Xqh, _Float16* __restrict__ Xth,
    float* __restrict__ sqa, float* __restrict__ sqb) {
  __shared__ float red[256];
  const int row = blockIdx.x;
  const int t = threadIdx.x;
  const float* src;
  _Float16* dst;
  float* nrm;
  int r;
  if (row < M_DIM) {
    r = row;
    src = Xq + (size_t)r * D_DIM;
    dst = Xqh + (size_t)r * D_DIM;
    nrm = sqa;
  } else {
    r = row - M_DIM;
    src = Xt + (size_t)r * D_DIM;
    dst = Xth + (size_t)r * D_DIM;
    nrm = sqb;
  }
  float v = src[t];
  dst[t] = (_Float16)v;
  red[t] = v * v;
  __syncthreads();
#pragma unroll
  for (int s = 128; s > 0; s >>= 1) {
    if (t < s) red[t] += red[t + s];
    __syncthreads();
  }
  if (t == 0) nrm[r] = red[0];
}

// ---------------------------------------------------------------------------
// Kernel 2: fused WMMA Gram tiles -> RBF -> alpha-weighted row reduction.
// grid = (M/BM, NSPLIT), block = 256 (8 waves).
// Each wave owns 32 M-rows (two register-resident A tiles) so every B
// operand load feeds TWO independent WMMA accumulator chains.
// ---------------------------------------------------------------------------
__global__ __launch_bounds__(256) void rbf_wmma_kernel(
    const _Float16* __restrict__ Xqh, const _Float16* __restrict__ Xth,
    const float* __restrict__ sqa, const float* __restrict__ sqb,
    const float* __restrict__ alpha, float* __restrict__ partial) {
  const int lane = threadIdx.x & 31;
  const int wave = threadIdx.x >> 5;
  const int rowBase = blockIdx.x * BM + wave * 32;
  const int nStart = blockIdx.y * NCHUNK;
  const bool hiHalf = lane >= 16;
  const int l16 = lane & 15;

  // --- A operands (16x32 f16, M x K) held in registers for all of K=256.
  // ISA layout: lanes 0-15 row=l16 carry K {k0..k0+7, k0+16..k0+23};
  //             lanes 16-31 row=l16 carry K {k0+8..k0+15, k0+24..k0+31}.
  const int ac = hiHalf ? 8 : 0;
  const _Float16* arow0 = Xqh + (size_t)(rowBase + l16) * D_DIM;
  const _Float16* arow1 = arow0 + (size_t)16 * D_DIM;
  v16h A0[8], A1[8];
#pragma unroll
  for (int ks = 0; ks < 8; ++ks) {
    const _Float16* p0 = arow0 + ks * 32 + ac;
    v8h lo0 = *(const v8h*)(p0);
    v8h hi0 = *(const v8h*)(p0 + 16);
    A0[ks] = __builtin_shufflevector(lo0, hi0, 0, 1, 2, 3, 4, 5, 6, 7,
                                     8, 9, 10, 11, 12, 13, 14, 15);
    const _Float16* p1 = arow1 + ks * 32 + ac;
    v8h lo1 = *(const v8h*)(p1);
    v8h hi1 = *(const v8h*)(p1 + 16);
    A1[ks] = __builtin_shufflevector(lo1, hi1, 0, 1, 2, 3, 4, 5, 6, 7,
                                     8, 9, 10, 11, 12, 13, 14, 15);
  }

  // C/D layout: lanes 0-15 VGPR r -> row base+r; lanes 16-31 -> base+8+r.
  const int rb0 = rowBase + (hiHalf ? 8 : 0);
  const int rb1 = rb0 + 16;
  float sqa0[8], sqa1[8];
#pragma unroll
  for (int r = 0; r < 8; ++r) {
    sqa0[r] = sqa[rb0 + r];
    sqa1[r] = sqa[rb1 + r];
  }

  float acc0[8], acc1[8];
#pragma unroll
  for (int r = 0; r < 8; ++r) { acc0[r] = 0.0f; acc1[r] = 0.0f; }

  // --- B operand (32x16 f16, K x N): lane n holds column n = l16,
  // lanes 0-15 K {k0..k0+7, k0+16..k0+23}, lanes 16-31 K {k0+8..k0+15, k0+24..k0+31}.
  const int bko = hiHalf ? 8 : 0;

  for (int n0 = nStart; n0 < nStart + NCHUNK; n0 += 16) {
    const _Float16* brow = Xth + (size_t)(n0 + l16) * D_DIM + bko;

    // Speculative prefetch of next B tile into the WGP cache
    // (lowers to global_prefetch_b8; dropped harmlessly at chunk end).
    __builtin_prefetch(brow + 16 * D_DIM, 0, 1);
    __builtin_prefetch(brow + 16 * D_DIM + 128, 0, 1);

    v8f c0 = {};
    v8f c1 = {};
#pragma unroll
    for (int ks = 0; ks < 8; ++ks) {
      v8h lo = *(const v8h*)(brow + ks * 32);
      v8h hi = *(const v8h*)(brow + ks * 32 + 16);
      v16h B = __builtin_shufflevector(lo, hi, 0, 1, 2, 3, 4, 5, 6, 7,
                                       8, 9, 10, 11, 12, 13, 14, 15);
      // (neg_a, A, neg_b, B, c_mod, C, reuse_a, reuse_b)
      c0 = __builtin_amdgcn_wmma_f32_16x16x32_f16(false, A0[ks], false, B,
                                                  (short)0, c0, false, false);
      c1 = __builtin_amdgcn_wmma_f32_16x16x32_f16(false, A1[ks], false, B,
                                                  (short)0, c1, false, false);
    }
    const float sqb_n = sqb[n0 + l16];
    const float al = alpha[n0 + l16];
#pragma unroll
    for (int r = 0; r < 8; ++r) {
      float d20 = sqa0[r] + sqb_n - 2.0f * c0[r];
      float d21 = sqa1[r] + sqb_n - 2.0f * c1[r];
      d20 = fmaxf(d20, 0.0f);
      d21 = fmaxf(d21, 0.0f);
      acc0[r] += __expf(-GAMMA_F * d20) * al;
      acc1[r] += __expf(-GAMMA_F * d21) * al;
    }
  }

  // Reduce across the 16 lanes of each half (fixed order -> deterministic).
  float* pout = partial + (size_t)blockIdx.y * M_DIM;
#pragma unroll
  for (int r = 0; r < 8; ++r) {
    float v0 = acc0[r];
    v0 += __shfl_xor(v0, 1, 32);
    v0 += __shfl_xor(v0, 2, 32);
    v0 += __shfl_xor(v0, 4, 32);
    v0 += __shfl_xor(v0, 8, 32);
    float v1 = acc1[r];
    v1 += __shfl_xor(v1, 1, 32);
    v1 += __shfl_xor(v1, 2, 32);
    v1 += __shfl_xor(v1, 4, 32);
    v1 += __shfl_xor(v1, 8, 32);
    if (l16 == 0) {
      pout[rb0 + r] = v0;
      pout[rb1 + r] = v1;
    }
  }
}

// ---------------------------------------------------------------------------
// Kernel 3: deterministic reduction of NSPLIT partials per output row.
// ---------------------------------------------------------------------------
__global__ __launch_bounds__(256) void reduce_kernel(
    const float* __restrict__ partial, float* __restrict__ out) {
  const int m = blockIdx.x * 256 + threadIdx.x;
  float s = 0.0f;
#pragma unroll
  for (int i = 0; i < NSPLIT; ++i) s += partial[(size_t)i * M_DIM + m];
  out[m] = s;
}

extern "C" void kernel_launch(void* const* d_in, const int* in_sizes, int n_in,
                              void* d_out, int out_size, void* d_ws, size_t ws_size,
                              hipStream_t stream) {
  const float* Xq = (const float*)d_in[0];     // [M, D] fp32
  const float* Xt = (const float*)d_in[1];     // [N, D] fp32
  const float* alpha = (const float*)d_in[2];  // [N, 1] fp32
  float* out = (float*)d_out;                  // [M] fp32

  char* w = (char*)d_ws;
  _Float16* Xqh = (_Float16*)w;                                        // 4 MB
  _Float16* Xth = (_Float16*)(w + (size_t)M_DIM * D_DIM * 2);          // 4 MB
  float* sqa = (float*)(w + (size_t)(M_DIM + N_DIM) * D_DIM * 2);      // 32 KB
  float* sqb = sqa + M_DIM;                                            // 32 KB
  float* partial = sqb + N_DIM;                                        // 512 KB

  cvt_norm_kernel<<<M_DIM + N_DIM, 256, 0, stream>>>(Xq, Xt, Xqh, Xth, sqa, sqb);
  rbf_wmma_kernel<<<dim3(M_DIM / BM, NSPLIT), 256, 0, stream>>>(
      Xqh, Xth, sqa, sqb, alpha, partial);
  reduce_kernel<<<M_DIM / 256, 256, 0, stream>>>(partial, out);
}
